// SemanticLSTM_38147899523393
// MI455X (gfx1250) — matrix operand order repaired and checked by reading, not compiled, not run on hardware
//
#include <hip/hip_runtime.h>
#include <hip/hip_bf16.h>

// ---------------------------------------------------------------------------
// Semantic LSTM captioner for MI455X (gfx1250, wave32, WMMA bf16 path).
//   B=128, 79 steps, E=I=H=512, S=300, C=2048, V=10000.  ~190 GFLOP total.
//  - GEMM weights pre-swizzled once into WMMA-B fragment order (bf16):
//    B fragment = 2x global_load_b128 per lane, zero conversion in hot loops
//  - A tiles staged in LDS (bf16, stride 520 -> conflict-free ds_load_b128);
//    each A fragment is loaded ONCE and feeds two WMMAs (two column tiles)
//  - cx (input path) hoisted out of the recurrence as one batched dual-GEMM
//  - vocab projection = one big parallel GEMM, non-temporal stores (404 MB)
// ---------------------------------------------------------------------------

#define B_   128
#define Tm1  79
#define TT   80
#define V_   10000
#define D_   512          // E = I = H
#define S_   300
#define C_   2048
#define R_   (B_*Tm1)     // 10112 rows = 632 tiles of 16
#define LDA_ 520          // LDS row stride in ushorts (16B pad: bank rotation)

typedef __attribute__((ext_vector_type(16))) __bf16 v16bf;
typedef __attribute__((ext_vector_type(8)))  float  v8f;

__device__ __forceinline__ unsigned short f32_to_bf16(float f){
  unsigned u = __float_as_uint(f);
  u += 0x7FFFu + ((u >> 16) & 1u);          // round-to-nearest-even
  return (unsigned short)(u >> 16);
}

union FragU { v16bf v; uint4 q[2]; };

// A fragment (16x32 bf16) from row-major LDS tile (ushort, stride LDA_).
// element e -> k = k0 + half*16 + e  => one contiguous
// 16-ushort run -> two 16B LDS loads per lane.
__device__ __forceinline__ v16bf frag_A_lds(const unsigned short* tile,
                                            int k0, int lane){
  const int half = lane >> 4, m = lane & 15;
  const uint4* p = (const uint4*)(tile + m * LDA_ + k0 + half * 16);
  FragU u; u.q[0] = p[0]; u.q[1] = p[1];
  return u.v;
}
// B fragment from weight pre-swizzled into fragment order: tile (kt,nt) holds
// 32 lanes x 16 bf16 contiguous -> two global_load_b128 per lane.
__device__ __forceinline__ v16bf frag_B_swz(const unsigned short* Wz, int ntN,
                                            int kt, int nt, int lane){
  const uint4* p = (const uint4*)(Wz + (((size_t)kt * ntN + nt) * 32 + lane) * 16);
  FragU u; u.q[0] = p[0]; u.q[1] = p[1];
  return u.v;
}
__device__ __forceinline__ v8f wmma_bf16(v16bf a, v16bf b, v8f c){
  return __builtin_amdgcn_wmma_f32_16x16x32_bf16(false, a, false, b, (short)0, c,
                                                 false, false);
}

// ---------------------------------------------------------------------------
__global__ void k_init_state(float* ht, float* ct){
  int i = blockIdx.x * blockDim.x + threadIdx.x;
  if (i < B_ * D_){ ht[i] = 0.f; ct[i] = 0.f; }
}

// f32 (K x N) -> bf16 in WMMA-B fragment order. blockIdx.y = batched matrix.
__global__ void k_swizzle(const float* __restrict__ src,
                          unsigned short* __restrict__ dst, int K, int N){
  int i = blockIdx.x * blockDim.x + threadIdx.x;
  if (i >= K * N) return;
  size_t base = (size_t)blockIdx.y * K * N;
  int k = i / N, n = i - k * N;
  int kt = k >> 5, nt = n >> 4;
  int half = (k >> 4) & 1, e = k & 15, ln = half * 16 + (n & 15);
  size_t d = (((size_t)kt * (N >> 4) + nt) * 32 + ln) * 16 + e;
  dst[base + d] = f32_to_bf16(src[base + i]);
}

// bx/bh/bv = einsum('bs,gsi->gbi'); one-time, fp32 VALU.
__global__ void k_semantic_b(const float* __restrict__ sem,
                             const float* __restrict__ Wbx,
                             const float* __restrict__ Wbh,
                             const float* __restrict__ Wbv,
                             float* bx, float* bh, float* bv){
  int idx = blockIdx.x * blockDim.x + threadIdx.x;
  if (idx >= 4 * B_ * D_) return;
  int g = idx >> 16, b = (idx >> 9) & 127, i = idx & 511;
  const float* s  = sem + b * S_;
  const float* wx = Wbx + (size_t)g * S_ * D_ + i;
  const float* wh = Wbh + (size_t)g * S_ * D_ + i;
  const float* wv = Wbv + (size_t)g * S_ * D_ + i;
  float ax = 0.f, ah = 0.f, av = 0.f;
  for (int k = 0; k < S_; ++k){
    float sv = s[k];
    ax += sv * wx[(size_t)k * D_];
    ah += sv * wh[(size_t)k * D_];
    av += sv * wv[(size_t)k * D_];
  }
  bx[idx] = ax; bh[idx] = ah; bv[idx] = av;
}

// av = einsum('bc,gci->gbi'); one-time.
__global__ void k_av(const float* __restrict__ cnn, const float* __restrict__ Wav,
                     float* av){
  int idx = blockIdx.x * blockDim.x + threadIdx.x;
  if (idx >= 4 * B_ * D_) return;
  int g = idx >> 16, b = (idx >> 9) & 127, i = idx & 511;
  const float* c = cnn + (size_t)b * C_;
  const float* w = Wav + (size_t)g * C_ * D_ + i;
  float acc = 0.f;
  for (int k = 0; k < C_; ++k) acc += c[k] * w[(size_t)k * D_];
  av[idx] = acc;
}

// cv = (av*bv) @ Wc_v. grid = 4 gates x 8 batch tiles, 512 threads.
__global__ void k_cv(const float* __restrict__ av, const float* __restrict__ bv,
                     const unsigned short* __restrict__ WzCv, float* cv){
  __shared__ unsigned short sU[16 * LDA_];
  int g = blockIdx.x >> 3, b0 = (blockIdx.x & 7) * 16;
  int tid = threadIdx.x;
  for (int i = tid; i < 16 * D_; i += 512){
    int row = i >> 9, k = i & 511;
    size_t off = ((size_t)g * B_ + b0 + row) * D_ + k;
    sU[row * LDA_ + k] = f32_to_bf16(av[off] * bv[off]);
  }
  __syncthreads();
  int wave = tid >> 5, lane = tid & 31, half = lane >> 4, n = lane & 15;
  const unsigned short* Wz = WzCv + (size_t)g * D_ * D_;
  int nt0 = wave * 2;
  v8f acc[2] = {};
#pragma unroll 4
  for (int kc = 0; kc < 16; ++kc){
    v16bf a = frag_A_lds(sU, kc * 32, lane);        // one A, two WMMAs
    acc[0] = wmma_bf16(a, frag_B_swz(Wz, 32, kc, nt0,     lane), acc[0]);
    acc[1] = wmma_bf16(a, frag_B_swz(Wz, 32, kc, nt0 + 1, lane), acc[1]);
  }
#pragma unroll
  for (int w = 0; w < 2; ++w)
#pragma unroll
    for (int r = 0; r < 8; ++r){
      int m = r + half * 8;
      cv[((size_t)g * B_ + b0 + m) * D_ + (nt0 + w) * 16 + n] = acc[w][r];
    }
}

// cx_all[g, b*79+t, :] = ((emb @ Wa_x[g]) * bx[g]) @ Wc_x[g]; fused dual GEMM
// per 16-row tile with embedding gather into the A tile. grid = 4*632, 512 thr.
__global__ void k_cx_all(const int* __restrict__ captions,
                         const float* __restrict__ embed,
                         const unsigned short* __restrict__ WzAx,
                         const unsigned short* __restrict__ WzCx,
                         const float* __restrict__ bx, float* cx_all){
  __shared__ unsigned short sA[16 * LDA_];
  __shared__ unsigned short sU[16 * LDA_];
  int g = blockIdx.x / 632;
  int r0 = (blockIdx.x % 632) * 16;
  int tid = threadIdx.x;
  for (int i = tid; i < 16 * D_; i += 512){
    int row = i >> 9, k = i & 511;
    int r = r0 + row, b = r / Tm1, t = r - b * Tm1;
    int cap = captions[b * TT + t];
    sA[row * LDA_ + k] = f32_to_bf16(embed[(size_t)cap * D_ + k]);
  }
  __syncthreads();
  int wave = tid >> 5, lane = tid & 31, half = lane >> 4, n = lane & 15;
  const unsigned short* Wa = WzAx + (size_t)g * D_ * D_;
  const unsigned short* Wc = WzCx + (size_t)g * D_ * D_;
  int nt0 = wave * 2;
  {                                                  // stage 1: ax*bx -> sU
    v8f acc[2] = {};
#pragma unroll 4
    for (int kc = 0; kc < 16; ++kc){
      v16bf a = frag_A_lds(sA, kc * 32, lane);
      acc[0] = wmma_bf16(a, frag_B_swz(Wa, 32, kc, nt0,     lane), acc[0]);
      acc[1] = wmma_bf16(a, frag_B_swz(Wa, 32, kc, nt0 + 1, lane), acc[1]);
    }
#pragma unroll
    for (int w = 0; w < 2; ++w)
#pragma unroll
      for (int r = 0; r < 8; ++r){
        int m = r + half * 8, row = r0 + m, b = row / Tm1;
        int h = (nt0 + w) * 16 + n;
        float v = acc[w][r] * bx[((size_t)g * B_ + b) * D_ + h];
        sU[m * LDA_ + h] = f32_to_bf16(v);
      }
  }
  __syncthreads();
  {                                                  // stage 2: sU @ Wc_x
    v8f acc[2] = {};
#pragma unroll 4
    for (int kc = 0; kc < 16; ++kc){
      v16bf a = frag_A_lds(sU, kc * 32, lane);
      acc[0] = wmma_bf16(a, frag_B_swz(Wc, 32, kc, nt0,     lane), acc[0]);
      acc[1] = wmma_bf16(a, frag_B_swz(Wc, 32, kc, nt0 + 1, lane), acc[1]);
    }
#pragma unroll
    for (int w = 0; w < 2; ++w)
#pragma unroll
      for (int r = 0; r < 8; ++r){
        int m = r + half * 8;
        cx_all[((size_t)g * R_ + r0 + m) * D_ + (nt0 + w) * 16 + n] = acc[w][r];
      }
  }
}

// One recurrent step, all 4 gates fused through LDS. grid = 8, block 512.
__global__ void k_step(int t, const unsigned short* __restrict__ WzAh,
                       const unsigned short* __restrict__ WzCh,
                       const float* __restrict__ bh,
                       const float* __restrict__ cx_all,
                       const float* __restrict__ cv,
                       const float* __restrict__ bias,
                       float* ht, float* ct, unsigned short* h_all){
  __shared__ unsigned short sH[16 * LDA_];
  __shared__ unsigned short sU[16 * LDA_];
  int b0 = blockIdx.x * 16;
  int tid = threadIdx.x;
  for (int i = tid; i < 16 * D_; i += 512){
    int row = i >> 9, k = i & 511;
    sH[row * LDA_ + k] = f32_to_bf16(ht[(b0 + row) * D_ + k]);
  }
  __syncthreads();
  int wave = tid >> 5, lane = tid & 31, half = lane >> 4, n = lane & 15;
  int nt0 = wave * 2;
  v8f zero = {};
  v8f acc2[2][4];
#pragma unroll
  for (int w = 0; w < 2; ++w)
#pragma unroll
    for (int g = 0; g < 4; ++g) acc2[w][g] = zero;

  for (int g = 0; g < 4; ++g){
    const unsigned short* Wa = WzAh + (size_t)g * D_ * D_;
    {                                               // ah = ht @ Wa_h[g], *bh
      v8f acc[2] = {};
#pragma unroll 4
      for (int kc = 0; kc < 16; ++kc){
        v16bf a = frag_A_lds(sH, kc * 32, lane);
        acc[0] = wmma_bf16(a, frag_B_swz(Wa, 32, kc, nt0,     lane), acc[0]);
        acc[1] = wmma_bf16(a, frag_B_swz(Wa, 32, kc, nt0 + 1, lane), acc[1]);
      }
#pragma unroll
      for (int w = 0; w < 2; ++w)
#pragma unroll
        for (int r = 0; r < 8; ++r){
          int m = r + half * 8, h = (nt0 + w) * 16 + n;
          float v = acc[w][r] * bh[((size_t)g * B_ + b0 + m) * D_ + h];
          sU[m * LDA_ + h] = f32_to_bf16(v);
        }
    }
    __syncthreads();
    const unsigned short* Wc = WzCh + (size_t)g * D_ * D_;
#pragma unroll 4
    for (int kc = 0; kc < 16; ++kc){                // ch partial into acc2
      v16bf a = frag_A_lds(sU, kc * 32, lane);
      acc2[0][g] = wmma_bf16(a, frag_B_swz(Wc, 32, kc, nt0,     lane), acc2[0][g]);
      acc2[1][g] = wmma_bf16(a, frag_B_swz(Wc, 32, kc, nt0 + 1, lane), acc2[1][g]);
    }
    __syncthreads();
  }
  // gates + cell update
  for (int w = 0; w < 2; ++w){
    int n0 = (nt0 + w) * 16;
#pragma unroll
    for (int r = 0; r < 8; ++r){
      int m = r + half * 8;
      int b = b0 + m, h = n0 + n;
      size_t rrow = (size_t)b * Tm1 + t;
      float gi = acc2[w][0][r] + cx_all[((size_t)0 * R_ + rrow) * D_ + h]
               + cv[((size_t)0 * B_ + b) * D_ + h] + bias[0 * D_ + h];
      float gf = acc2[w][1][r] + cx_all[((size_t)1 * R_ + rrow) * D_ + h]
               + cv[((size_t)1 * B_ + b) * D_ + h] + bias[1 * D_ + h];
      float go = acc2[w][2][r] + cx_all[((size_t)2 * R_ + rrow) * D_ + h]
               + cv[((size_t)2 * B_ + b) * D_ + h] + bias[2 * D_ + h];
      float gg = acc2[w][3][r] + cx_all[((size_t)3 * R_ + rrow) * D_ + h]
               + cv[((size_t)3 * B_ + b) * D_ + h] + bias[3 * D_ + h];
      float i_s = 1.f / (1.f + __expf(-gi));
      float f_s = 1.f / (1.f + __expf(-gf));
      float o_s = 1.f / (1.f + __expf(-go));
      float g_t = tanhf(gg);
      float cn  = f_s * ct[b * D_ + h] + i_s * g_t;
      ct[b * D_ + h] = cn;
      float hn = o_s * tanhf(cn);
      ht[b * D_ + h] = hn;
      h_all[rrow * D_ + h] = f32_to_bf16(hn);
    }
  }
}

// logits: (10112x512)@(512x10000)+bout; NT stores (don't pollute L2).
// grid = 632 row-tiles x 20 col-tiles of 512; block 512 (16 waves x 2 nt).
__global__ void k_logits(const unsigned short* __restrict__ h_all,
                         const unsigned short* __restrict__ WzOut,
                         const float* __restrict__ bout, float* out){
  __shared__ unsigned short sA[16 * LDA_];
  int rt = blockIdx.x / 20, cb = blockIdx.x % 20;
  int r0 = rt * 16, c0 = cb * 512;
  int tid = threadIdx.x;
  // vectorized copy: h_all rows (bf16) -> padded LDS tile, 8 ushorts at a time
  for (int j = tid; j < 16 * D_ / 8; j += 512){
    int row = j >> 6, col = (j & 63) * 8;
    *(uint4*)(sA + row * LDA_ + col) =
        *(const uint4*)(h_all + (size_t)(r0 + row) * D_ + col);
  }
  __syncthreads();
  int wave = tid >> 5, lane = tid & 31, half = lane >> 4, n = lane & 15;
  int nt0 = (c0 >> 4) + wave * 2;
  bool ok0 = (nt0    ) * 16 < V_;                  // wave-uniform guards
  bool ok1 = (nt0 + 1) * 16 < V_;
  if (!ok0) return;
  v8f acc[2] = {};
  if (ok1){
#pragma unroll 4
    for (int kc = 0; kc < 16; ++kc){
      v16bf a = frag_A_lds(sA, kc * 32, lane);
      acc[0] = wmma_bf16(a, frag_B_swz(WzOut, V_ / 16, kc, nt0,     lane), acc[0]);
      acc[1] = wmma_bf16(a, frag_B_swz(WzOut, V_ / 16, kc, nt0 + 1, lane), acc[1]);
    }
  } else {
#pragma unroll 4
    for (int kc = 0; kc < 16; ++kc){
      v16bf a = frag_A_lds(sA, kc * 32, lane);
      acc[0] = wmma_bf16(a, frag_B_swz(WzOut, V_ / 16, kc, nt0, lane), acc[0]);
    }
  }
  int nw = ok1 ? 2 : 1;
  for (int w = 0; w < nw; ++w){
    int n0 = (nt0 + w) * 16;
#pragma unroll
    for (int r = 0; r < 8; ++r){
      int m = r + half * 8;
      __builtin_nontemporal_store(acc[w][r] + bout[n0 + n],
                                  &out[(size_t)(r0 + m) * V_ + n0 + n]);
    }
  }
}

// ---------------------------------------------------------------------------
extern "C" void kernel_launch(void* const* d_in, const int* in_sizes, int n_in,
                              void* d_out, int out_size, void* d_ws, size_t ws_size,
                              hipStream_t stream){
  (void)in_sizes; (void)n_in; (void)out_size; (void)ws_size;
  const int*   captions = (const int*)  d_in[0];
  const float* cnn      = (const float*)d_in[1];
  const float* sem      = (const float*)d_in[2];
  const float* Wa_x     = (const float*)d_in[3];
  const float* Wb_x     = (const float*)d_in[4];
  const float* Wc_x     = (const float*)d_in[5];
  const float* Wa_h     = (const float*)d_in[6];
  const float* Wb_h     = (const float*)d_in[7];
  const float* Wc_h     = (const float*)d_in[8];
  const float* Wa_v     = (const float*)d_in[9];
  const float* Wb_v     = (const float*)d_in[10];
  const float* Wc_v     = (const float*)d_in[11];
  const float* bias     = (const float*)d_in[12];
  const float* embed    = (const float*)d_in[13];
  const float* Wout     = (const float*)d_in[14];
  const float* bout     = (const float*)d_in[15];
  float* out = (float*)d_out;

  // workspace layout (~117 MB)
  char* ws = (char*)d_ws; size_t off = 0;
  float* cx_all = (float*)(ws + off); off += (size_t)4 * R_ * D_ * 4;   // 82.8MB
  float* cv     = (float*)(ws + off); off += (size_t)4 * B_ * D_ * 4;
  float* bx     = (float*)(ws + off); off += (size_t)4 * B_ * D_ * 4;
  float* bh     = (float*)(ws + off); off += (size_t)4 * B_ * D_ * 4;
  float* bv     = (float*)(ws + off); off += (size_t)4 * B_ * D_ * 4;
  float* av     = (float*)(ws + off); off += (size_t)4 * B_ * D_ * 4;
  float* ht     = (float*)(ws + off); off += (size_t)B_ * D_ * 4;
  float* ct     = (float*)(ws + off); off += (size_t)B_ * D_ * 4;
  unsigned short* h_all = (unsigned short*)(ws + off); off += (size_t)R_ * D_ * 2;
  unsigned short* wzAx  = (unsigned short*)(ws + off); off += (size_t)4 * D_ * D_ * 2;
  unsigned short* wzCx  = (unsigned short*)(ws + off); off += (size_t)4 * D_ * D_ * 2;
  unsigned short* wzAh  = (unsigned short*)(ws + off); off += (size_t)4 * D_ * D_ * 2;
  unsigned short* wzCh  = (unsigned short*)(ws + off); off += (size_t)4 * D_ * D_ * 2;
  unsigned short* wzCv  = (unsigned short*)(ws + off); off += (size_t)4 * D_ * D_ * 2;
  unsigned short* wzOut = (unsigned short*)(ws + off); off += (size_t)D_ * V_ * 2;

  k_init_state<<<(B_ * D_ + 255) / 256, 256, 0, stream>>>(ht, ct);
  dim3 gsw((D_ * D_ + 255) / 256, 4);
  k_swizzle<<<gsw, 256, 0, stream>>>(Wa_x, wzAx, D_, D_);
  k_swizzle<<<gsw, 256, 0, stream>>>(Wc_x, wzCx, D_, D_);
  k_swizzle<<<gsw, 256, 0, stream>>>(Wa_h, wzAh, D_, D_);
  k_swizzle<<<gsw, 256, 0, stream>>>(Wc_h, wzCh, D_, D_);
  k_swizzle<<<gsw, 256, 0, stream>>>(Wc_v, wzCv, D_, D_);
  k_swizzle<<<dim3((D_ * V_ + 255) / 256, 1), 256, 0, stream>>>(Wout, wzOut, D_, V_);
  k_semantic_b<<<(4 * B_ * D_) / 256, 256, 0, stream>>>(sem, Wb_x, Wb_h, Wb_v,
                                                        bx, bh, bv);
  k_av<<<(4 * B_ * D_) / 256, 256, 0, stream>>>(cnn, Wa_v, av);
  k_cv<<<32, 512, 0, stream>>>(av, bv, wzCv, cv);
  k_cx_all<<<4 * 632, 512, 0, stream>>>(captions, embed, wzAx, wzCx, bx, cx_all);
  for (int t = 0; t < Tm1; ++t)
    k_step<<<8, 512, 0, stream>>>(t, wzAh, wzCh, bh, cx_all, cv, bias,
                                  ht, ct, h_all);
  k_logits<<<632 * 20, 512, 0, stream>>>(h_all, wzOut, bout, out);
}
